// local_policy_88313117540882
// MI455X (gfx1250) — matrix operand order repaired and checked by reading, not compiled, not run on hardware
//
#include <hip/hip_runtime.h>
#include <hip/hip_bf16.h>

// ---------------------------------------------------------------------------
// Problem constants (match the JAX reference)
// ---------------------------------------------------------------------------
#define BB     8
#define NN     2000
#define VV     2000
#define KK     100
#define EMB    128
#define EMB2   256          // 2*EMB
#define ROWS   (BB * NN)    // 16000
#define KPAD   224          // 2K+1 = 201 padded to 7 * 32
#define NPAD4  112          // K=100 padded to 7 * 16 output cols of layer 4
#define PENALTY (-1e9f)
#define EPSI    (1e-5f)

typedef __attribute__((ext_vector_type(16))) _Float16 v16h;
typedef __attribute__((ext_vector_type(8)))  _Float16 v8h;
typedef __attribute__((ext_vector_type(8)))  float    v8f;

// ---------------------------------------------------------------------------
// Fragment load helpers (CDNA5 wave32 WMMA 16x16x32 f16 layouts, ISA 7.12.2)
//   A 16x32 f16: lanes 0-15 -> M=lane, halfs = K[0..7] ++ K[16..23]
//                lanes 16-31 -> M=lane-16, halfs = K[8..15] ++ K[24..31]
//   B 32x16 f16: lane<16 -> col=lane,     halfs = K[0..7] ++ K[16..23]   (of the k-tile)
//                lane>=16 -> col=lane-16, halfs = K[8..15] ++ K[24..31]
// Weights are pre-repacked so each lane's 16 B-halfs are contiguous (32B).
// ---------------------------------------------------------------------------
__device__ __forceinline__ v16h load_a_frag(const _Float16* base, int ld,
                                            int kt, int lane) {
  const int m  = lane & 15;
  const int k0 = kt * 32 + ((lane >> 4) << 3);
  const _Float16* p = base + m * ld + k0;
  v8h lo = *reinterpret_cast<const v8h*>(p);
  v8h hi = *reinterpret_cast<const v8h*>(p + 16);
  v16h a;
#pragma unroll
  for (int j = 0; j < 8; ++j) { a[j] = lo[j]; a[j + 8] = hi[j]; }
  return a;
}

__device__ __forceinline__ v16h load_b_frag(const _Float16* packed,
                                            int numKt, int nt, int kt, int lane) {
  const _Float16* p = packed + (((size_t)(nt * numKt + kt) * 32 + lane) << 4);
  return *reinterpret_cast<const v16h*>(p);
}

__device__ __forceinline__ v8f wmma_f16(v16h a, v16h b, v8f c) {
  return __builtin_amdgcn_wmma_f32_16x16x32_f16(false, a, false, b,
                                                (short)0, c, false, false);
}

// ---------------------------------------------------------------------------
// Kernel 0: fill output with PENALTY (vectorized)
// ---------------------------------------------------------------------------
__global__ void fill_penalty_kernel(float4* out, int n4) {
  int i = blockIdx.x * blockDim.x + threadIdx.x;
  if (i < n4) out[i] = make_float4(PENALTY, PENALTY, PENALTY, PENALTY);
}

// ---------------------------------------------------------------------------
// Kernel 1: repack a [Kdim, Nout] row-major f32 weight into B-fragment f16
// layout: P[((nt*numKt + kt)*32 + lane)*16 + j]
//   lane<16 : kk = kt*32 + (j<8 ? j : j+8),     col = nt*16 + lane
//   lane>=16: kk = kt*32 + 8 + (j<8 ? j : j+8), col = nt*16 + lane - 16
// Zero-pads out-of-range K / N.
// ---------------------------------------------------------------------------
__global__ void repack_w_kernel(const float* W, _Float16* P,
                                int Kdim, int Nout, int numKt, int numNt) {
  int idx = blockIdx.x * blockDim.x + threadIdx.x;
  int total = numKt * numNt * 512;
  if (idx >= total) return;
  int j    = idx & 15;
  int lane = (idx >> 4) & 31;
  int tile = idx >> 9;
  int kt   = tile % numKt;
  int nt   = tile / numKt;
  int kk   = kt * 32 + ((lane >= 16) ? 8 : 0) + ((j < 8) ? j : j + 8);
  int col  = nt * 16 + (lane & 15);
  float v  = (kk < Kdim && col < Nout) ? W[(size_t)kk * Nout + col] : 0.0f;
  P[idx]   = (_Float16)v;
}

// ---------------------------------------------------------------------------
// Kernel 2: per-row top-K (ascending, tie-break lowest index) via u64 bitonic
// sort in LDS; emits normalized sorted_dist, indices, and the padded f16
// activation row xin[row][0:224] = [dist/max (100) | theta@idx (100) | scale | 0...]
// ---------------------------------------------------------------------------
__global__ void topk_build_kernel(const float* __restrict__ dist,
                                  const float* __restrict__ theta,
                                  const float* __restrict__ scale,
                                  float* __restrict__ sdist,
                                  int*   __restrict__ sidx,
                                  _Float16* __restrict__ xin) {
  __shared__ unsigned long long keys[2048];
  __shared__ float maxd_s;
  const int row = blockIdx.x;
  const int tid = threadIdx.x;
  const size_t rbase = (size_t)row * VV;

  for (int i = tid; i < 2048; i += 256) {
    if (i < VV) {
      unsigned bits = __float_as_uint(dist[rbase + i]);  // uniform [0,1): positive
      keys[i] = ((unsigned long long)bits << 32) | (unsigned)i;
    } else {
      keys[i] = ~0ull;
    }
  }
  __syncthreads();

  for (int k = 2; k <= 2048; k <<= 1) {
    for (int j = k >> 1; j > 0; j >>= 1) {
      for (int i = tid; i < 2048; i += 256) {
        int ixj = i ^ j;
        if (ixj > i) {
          bool up = ((i & k) == 0);
          unsigned long long a = keys[i], b = keys[ixj];
          if ((a > b) == up) { keys[i] = b; keys[ixj] = a; }
        }
      }
      __syncthreads();
    }
  }

  if (tid == 0) maxd_s = __uint_as_float((unsigned)(keys[KK - 1] >> 32));
  __syncthreads();
  const float inv_max = 1.0f / maxd_s;

  _Float16* xrow = xin + (size_t)row * KPAD;
  if (tid < KK) {
    unsigned long long kv = keys[tid];
    float v = __uint_as_float((unsigned)(kv >> 32)) * inv_max;  // normalized
    int   id = (int)(unsigned)(kv & 0xffffffffu);
    sdist[(size_t)row * KK + tid] = v;
    sidx [(size_t)row * KK + tid] = id;
    xrow[tid]      = (_Float16)v;
    xrow[KK + tid] = (_Float16)theta[rbase + id];
  }
  if (tid < (KPAD - 2 * KK)) {  // fill [200, 224)
    xrow[2 * KK + tid] = (tid == 0) ? (_Float16)scale[row] : (_Float16)0.0f;
  }
}

// ---------------------------------------------------------------------------
// Kernel 3: fused layer1+layer2.  16-row tile per block, 8 waves.
//   emb = relu(xin @ w1 + b1)   [16 x 128]  (wave w -> N-tile w)
//   h   = relu(emb @ w2 + b2)   [16 x 256]  (wave w -> N-tiles 2w, 2w+1)
// ---------------------------------------------------------------------------
__global__ void mlp12_kernel(const _Float16* __restrict__ xin,
                             const _Float16* __restrict__ pw1,
                             const float*    __restrict__ b1,
                             const _Float16* __restrict__ pw2,
                             const float*    __restrict__ b2,
                             _Float16* __restrict__ hout) {
  __shared__ _Float16 embs[16 * EMB];
  const int row0 = blockIdx.x * 16;
  const int wave = threadIdx.x >> 5;
  const int lane = threadIdx.x & 31;
  const int moff = (lane >> 4) << 3;     // 0 or 8
  const int ncol = lane & 15;

  // ---- layer 1: Kdim 224 (7 k-tiles), N = 128 (8 n-tiles, one per wave) ----
  {
    const int nt = wave;
    v8f acc = {};
#pragma unroll
    for (int kt = 0; kt < 7; ++kt) {
      v16h a = load_a_frag(xin + (size_t)row0 * KPAD, KPAD, kt, lane);
      v16h b = load_b_frag(pw1, 7, nt, kt, lane);
      acc = wmma_f16(a, b, acc);
    }
    const int n = nt * 16 + ncol;
    const float bias = b1[n];
#pragma unroll
    for (int r = 0; r < 8; ++r) {
      float v = acc[r] + bias;
      v = v > 0.0f ? v : 0.0f;
      embs[(moff + r) * EMB + n] = (_Float16)v;
    }
  }
  __syncthreads();

  // ---- layer 2: Kdim 128 (4 k-tiles), N = 256 (16 n-tiles, 2 per wave) ----
#pragma unroll
  for (int t = 0; t < 2; ++t) {
    const int nt = wave * 2 + t;
    v8f acc = {};
#pragma unroll
    for (int kt = 0; kt < 4; ++kt) {
      v16h a = load_a_frag(embs, EMB, kt, lane);
      v16h b = load_b_frag(pw2, 4, nt, kt, lane);
      acc = wmma_f16(a, b, acc);
    }
    const int n = nt * 16 + ncol;
    const float bias = b2[n];
#pragma unroll
    for (int r = 0; r < 8; ++r) {
      float v = acc[r] + bias;
      v = v > 0.0f ? v : 0.0f;
      hout[(size_t)(row0 + moff + r) * EMB2 + n] = (_Float16)v;
    }
  }
}

// ---------------------------------------------------------------------------
// Kernel 4: InstanceNorm stats over the node dim (per b, per channel).
// Precompute affine:  y = h * aScale + aShift
//   aScale = gamma * rsqrt(var + eps);  aShift = beta - mean * aScale
// ---------------------------------------------------------------------------
__global__ void inorm_stats_kernel(const _Float16* __restrict__ h,
                                   const float* __restrict__ gamma,
                                   const float* __restrict__ beta,
                                   float* __restrict__ aScale,
                                   float* __restrict__ aShift) {
  const int b = blockIdx.x;
  const int c = threadIdx.x;           // 256 channels
  float s = 0.0f, sq = 0.0f;
  const _Float16* p = h + ((size_t)b * NN) * EMB2 + c;
  for (int n = 0; n < NN; ++n) {
    float v = (float)p[(size_t)n * EMB2];
    s += v; sq += v * v;
  }
  const float mean = s * (1.0f / NN);
  float var = sq * (1.0f / NN) - mean * mean;
  var = var > 0.0f ? var : 0.0f;
  const float sc = gamma[c] * rsqrtf(var + EPSI);
  aScale[b * EMB2 + c] = sc;
  aShift[b * EMB2 + c] = beta[c] - mean * sc;
}

// ---------------------------------------------------------------------------
// Kernel 5: fused norm + layer3 + layer4 + (-sorted_dist) + scatter.
//   nh   = h * aScale + aShift           [16 x 256] (LDS, f16)
//   emb2 = relu(nh @ w3 + b3)            [16 x 128] (LDS, f16)
//   out  = emb2 @ w4 + b4 - sdist  -> scatter to d_out[b, n, idx]
// 2000 % 16 == 0 so a 16-row tile never straddles a batch.
// ---------------------------------------------------------------------------
__global__ void mlp34_scatter_kernel(const _Float16* __restrict__ h,
                                     const float* __restrict__ aScale,
                                     const float* __restrict__ aShift,
                                     const _Float16* __restrict__ pw3,
                                     const float*    __restrict__ b3,
                                     const _Float16* __restrict__ pw4,
                                     const float*    __restrict__ b4,
                                     const float* __restrict__ sdist,
                                     const int*   __restrict__ sidx,
                                     float* __restrict__ out) {
  __shared__ _Float16 nh[16 * EMB2];
  __shared__ _Float16 emb2[16 * EMB];
  const int row0 = blockIdx.x * 16;
  const int b    = row0 / NN;
  const int wave = threadIdx.x >> 5;
  const int lane = threadIdx.x & 31;
  const int moff = (lane >> 4) << 3;
  const int ncol = lane & 15;

  // apply InstanceNorm affine into LDS
  for (int t = threadIdx.x; t < 16 * EMB2; t += 256) {
    int r = t >> 8, c = t & 255;
    float v = (float)h[(size_t)(row0 + r) * EMB2 + c];
    nh[t] = (_Float16)(v * aScale[b * EMB2 + c] + aShift[b * EMB2 + c]);
  }
  __syncthreads();

  // ---- layer 3: Kdim 256 (8 k-tiles), N = 128 (8 n-tiles, 1 per wave) ----
  {
    const int nt = wave;
    v8f acc = {};
#pragma unroll
    for (int kt = 0; kt < 8; ++kt) {
      v16h a = load_a_frag(nh, EMB2, kt, lane);
      v16h bf = load_b_frag(pw3, 8, nt, kt, lane);
      acc = wmma_f16(a, bf, acc);
    }
    const int n = nt * 16 + ncol;
    const float bias = b3[n];
#pragma unroll
    for (int r = 0; r < 8; ++r) {
      float v = acc[r] + bias;
      v = v > 0.0f ? v : 0.0f;
      emb2[(moff + r) * EMB + n] = (_Float16)v;
    }
  }
  __syncthreads();

  // ---- layer 4: Kdim 128 (4 k-tiles), N = 112 (7 n-tiles, waves 0..6) ----
  if (wave < 7) {
    const int nt = wave;
    v8f acc = {};
#pragma unroll
    for (int kt = 0; kt < 4; ++kt) {
      v16h a = load_a_frag(emb2, EMB, kt, lane);
      v16h bf = load_b_frag(pw4, 4, nt, kt, lane);
      acc = wmma_f16(a, bf, acc);
    }
    const int col = nt * 16 + ncol;
    if (col < KK) {
      const float bias = b4[col];
#pragma unroll
      for (int r = 0; r < 8; ++r) {
        int g = row0 + moff + r;
        float v = acc[r] + bias - sdist[(size_t)g * KK + col];
        int id  = sidx[(size_t)g * KK + col];
        out[(size_t)g * VV + id] = v;
      }
    }
  }
}

// ---------------------------------------------------------------------------
// Host launcher
// ---------------------------------------------------------------------------
extern "C" void kernel_launch(void* const* d_in, const int* in_sizes, int n_in,
                              void* d_out, int out_size, void* d_ws, size_t ws_size,
                              hipStream_t stream) {
  const float* dist  = (const float*)d_in[0];
  const float* theta = (const float*)d_in[1];
  const float* scale = (const float*)d_in[2];
  const float* w1    = (const float*)d_in[3];
  const float* b1    = (const float*)d_in[4];
  const float* w2    = (const float*)d_in[5];
  const float* b2    = (const float*)d_in[6];
  const float* w3    = (const float*)d_in[7];
  const float* b3    = (const float*)d_in[8];
  const float* w4    = (const float*)d_in[9];
  const float* b4    = (const float*)d_in[10];
  const float* gamma = (const float*)d_in[11];
  const float* beta  = (const float*)d_in[12];
  float* out = (float*)d_out;

  char* ws = (char*)d_ws;
  size_t off = 0;
  auto take = [&](size_t bytes) -> char* {
    char* p = ws + off;
    off += (bytes + 255) & ~(size_t)255;
    return p;
  };
  _Float16* pw1  = (_Float16*)take((size_t)KPAD * EMB  * 2);
  _Float16* pw2  = (_Float16*)take((size_t)EMB  * EMB2 * 2);
  _Float16* pw3  = (_Float16*)take((size_t)EMB2 * EMB  * 2);
  _Float16* pw4  = (_Float16*)take((size_t)EMB  * NPAD4 * 2);
  _Float16* xin  = (_Float16*)take((size_t)ROWS * KPAD * 2);
  float*    sd   = (float*)   take((size_t)ROWS * KK * 4);
  int*      si   = (int*)     take((size_t)ROWS * KK * 4);
  _Float16* hbuf = (_Float16*)take((size_t)ROWS * EMB2 * 2);
  float*    aSc  = (float*)   take((size_t)BB * EMB2 * 4);
  float*    aSh  = (float*)   take((size_t)BB * EMB2 * 4);

  // 0) fill output with PENALTY
  {
    int n4 = (BB * NN * VV) / 4;                       // 8,000,000 float4
    fill_penalty_kernel<<<(n4 + 255) / 256, 256, 0, stream>>>((float4*)out, n4);
  }
  // 1) repack weights into WMMA B-fragment layout
  repack_w_kernel<<<(7 * 8  * 512 + 255) / 256, 256, 0, stream>>>(w1, pw1, 2 * KK + 1, EMB,  7, 8);
  repack_w_kernel<<<(4 * 16 * 512 + 255) / 256, 256, 0, stream>>>(w2, pw2, EMB,  EMB2, 4, 16);
  repack_w_kernel<<<(8 * 8  * 512 + 255) / 256, 256, 0, stream>>>(w3, pw3, EMB2, EMB,  8, 8);
  repack_w_kernel<<<(4 * 7  * 512 + 255) / 256, 256, 0, stream>>>(w4, pw4, EMB,  KK,   4, 7);
  // 2) top-k + activation build
  topk_build_kernel<<<ROWS, 256, 0, stream>>>(dist, theta, scale, sd, si, xin);
  // 3) layers 1+2
  mlp12_kernel<<<ROWS / 16, 256, 0, stream>>>(xin, pw1, b1, pw2, b2, hbuf);
  // 4) InstanceNorm stats
  inorm_stats_kernel<<<BB, EMB2, 0, stream>>>(hbuf, gamma, beta, aSc, aSh);
  // 5) norm + layers 3+4 + scatter
  mlp34_scatter_kernel<<<ROWS / 16, 256, 0, stream>>>(hbuf, aSc, aSh, pw3, b3,
                                                      pw4, b4, sd, si, out);
}
